// Attention_56178172231790
// MI455X (gfx1250) — matrix operand order repaired and checked by reading, not compiled, not run on hardware
//
#include <hip/hip_runtime.h>
#include <hip/hip_bf16.h>

// Talking-heads attention, fused flash-style pipeline for gfx1250 (MI455X).
// bf16 WMMA (v_wmma_f32_16x16x32_bf16) everywhere, f32 accumulation.
// GEMMs (K1/K3) use double-buffered GLOBAL_LOAD_ASYNC_TO_LDS_B128 staging
// (ASYNCcnt) to overlap memory with WMMA. Workspace: ~68 MB.

#define B_     8
#define N_     1024
#define DIM_   768
#define H_     12
#define DH_    64
#define INNER_ 768

typedef __attribute__((ext_vector_type(16))) __bf16 v16bf;
typedef __attribute__((ext_vector_type(8)))  __bf16 v8bf;
typedef __attribute__((ext_vector_type(8)))  float  v8f;

__device__ __forceinline__ unsigned short f2bf(float f) {
  unsigned u = __float_as_uint(f);
  u += 0x7fffu + ((u >> 16) & 1u);          // round-to-nearest-even
  return (unsigned short)(u >> 16);
}

__device__ __forceinline__ v8f vzero8() {
  v8f v;
  #pragma unroll
  for (int i = 0; i < 8; i++) v[i] = 0.0f;
  return v;
}

// Wave-relative LDS byte offset of a shared-memory pointer (HW adds LDS_BASE).
__device__ __forceinline__ unsigned lds_off(const void* p) {
  return (unsigned)(size_t)(__attribute__((address_space(3))) const char*)p;
}

// Async DMA: LDS[lds_byte_off] <= 16B at global address g.  Tracked by ASYNCcnt.
__device__ __forceinline__ void async_ld_b128(unsigned lds_byte_off, const void* g) {
  asm volatile("global_load_async_to_lds_b128 %0, %1, off"
               :: "v"(lds_byte_off), "v"((unsigned long long)(size_t)g)
               : "memory");
}

// A-matrix fragment (16x32 bf16, row-major source, leading dim `ld`).
// Lane L: row M = L&15, k0 = (L<16?0:8); elems 0..7 = K[k0..k0+7],
// elems 8..15 = K[k0+16..k0+23]  (ISA 16-bit A layout).
__device__ __forceinline__ v16bf load_a16(const __bf16* base, int ld, int lane) {
  int m  = lane & 15;
  int k0 = (lane >> 4) << 3;
  const __bf16* p = base + m * ld + k0;
  union { v16bf v; v8bf h[2]; } u;
  u.h[0] = *(const v8bf*)(p);
  u.h[1] = *(const v8bf*)(p + 16);
  return u.v;
}

// B-matrix fragment (32x16 bf16). Storage: element (k,n) at base[n*ld + k].
// Lane L: col N = L&15, k0 = (L<16?0:8); elems 0..7 = K[k0..k0+7], elems 8..15 = K[k0+16..k0+23].
__device__ __forceinline__ v16bf load_b16(const __bf16* base, int ld, int lane) {
  int n  = lane & 15;
  int kb = (lane >> 4) << 3;
  const __bf16* p = base + n * ld + kb;
  union { v16bf v; v8bf h[2]; } u;
  u.h[0] = *(const v8bf*)(p);
  u.h[1] = *(const v8bf*)(p + 16);
  return u.v;
}

__device__ __forceinline__ v8f wmma_bf16(v16bf a, v16bf b, v8f c) {
  return __builtin_amdgcn_wmma_f32_16x16x32_bf16(false, a, false, b,
                                                 (short)0, c, false, false);
}

// ---------------------------------------------------------------- prep ----
__global__ void k_cvt_f32_bf16(const float* __restrict__ src,
                               __bf16* __restrict__ dst, int n) {
  int i = blockIdx.x * blockDim.x + threadIdx.x;
  if (i < n) ((unsigned short*)dst)[i] = f2bf(src[i]);
}

// dst[j*rows + c] = bf16(src[c*src_ld + src_off + j])  (transpose + convert)
__global__ void k_transpose_cvt(const float* __restrict__ src,
                                __bf16* __restrict__ dst,
                                int rows, int cols, int src_ld, int src_off) {
  int i = blockIdx.x * blockDim.x + threadIdx.x;
  if (i >= rows * cols) return;
  int j = i / rows, c = i % rows;
  ((unsigned short*)dst)[(size_t)j * rows + c] =
      f2bf(src[(size_t)c * src_ld + src_off + j]);
}

// ------------------------------------------------------------- K1: QKV ----
// x[8192,768] bf16 @ {WqT|WkT|WvT}[768,768] (pre-transposed, B-ready).
// Block: 256 thr (8 waves), tile M=128 x N=64, K-chunks of 32, double-buffered
// async-to-LDS staging. q,k -> [B,H,N,DH]; v -> vT[B,H,DH,N]; q scaled DH^-0.5.
__global__ __launch_bounds__(256) void k_qkv(
    const __bf16* __restrict__ xb, const __bf16* __restrict__ WqT,
    const __bf16* __restrict__ WkT, const __bf16* __restrict__ WvT,
    __bf16* __restrict__ q, __bf16* __restrict__ k, __bf16* __restrict__ vT) {
  __shared__ __align__(16) __bf16 ldsX[2][128 * 40];
  __shared__ __align__(16) __bf16 ldsW[2][64 * 40];
  const int t = threadIdx.x, lane = t & 31, w = t >> 5;
  const int jg = blockIdx.x * 64;           // global output column [0,2304)
  const int region = jg / 768;              // 0=q 1=k 2=v
  const int jl0 = jg % 768;
  const int row0 = blockIdx.y * 128;        // global row [0,8192)
  const __bf16* WT = (region == 0) ? WqT : (region == 1) ? WkT : WvT;

  // per-thread staging assignment (fixed across K-chunks)
  const int xr = t >> 1, xc = (t & 1) * 16;   // 128 x 32 tile, 16 elems/thr
  const int wj = t >> 2, wc = (t & 3) * 8;    // 64 x 32 tile, 8 elems/thr
  const __bf16* xp = xb + (size_t)(row0 + xr) * DIM_ + xc;
  const __bf16* wp = WT + (size_t)(jl0 + wj) * DIM_ + wc;
  const unsigned lx[2] = { lds_off(&ldsX[0][xr * 40 + xc]),
                           lds_off(&ldsX[1][xr * 40 + xc]) };
  const unsigned lw[2] = { lds_off(&ldsW[0][wj * 40 + wc]),
                           lds_off(&ldsW[1][wj * 40 + wc]) };

  v8f acc[4];
  #pragma unroll
  for (int i = 0; i < 4; i++) acc[i] = vzero8();

  // prologue: stage chunk 0 into buffer 0
  async_ld_b128(lx[0],      xp);
  async_ld_b128(lx[0] + 16, xp + 8);
  async_ld_b128(lw[0],      wp);

  for (int kc = 0; kc < DIM_; kc += 32) {
    const int cur = (kc >> 5) & 1;
    if (kc + 32 < DIM_) {                     // stage next chunk into other buf
      const int nb = cur ^ 1;
      async_ld_b128(lx[nb],      xp + kc + 32);
      async_ld_b128(lx[nb] + 16, xp + kc + 40);
      async_ld_b128(lw[nb],      wp + kc + 32);
      asm volatile("s_wait_asynccnt 0x3" ::: "memory");  // cur's 3 ops done
    } else {
      asm volatile("s_wait_asynccnt 0x0" ::: "memory");
    }
    __syncthreads();                          // cur tile visible to all waves
    v16bf a = load_a16(&ldsX[cur][(w * 16) * 40], 40, lane);
    #pragma unroll
    for (int nb = 0; nb < 4; nb++) {
      v16bf b = load_b16(&ldsW[cur][(nb * 16) * 40], 40, lane);
      acc[nb] = wmma_bf16(a, b, acc[nb]);
    }
    __syncthreads();                          // reads done; cur may be restaged
  }

  const float scale = (region == 0) ? 0.125f : 1.0f;  // DH^-0.5 folded into q
  const int hi = lane >> 4, ln = lane & 15;
  #pragma unroll
  for (int nb = 0; nb < 4; nb++) {
    int jl = jl0 + nb * 16 + ln;
    int h = jl >> 6, d = jl & 63;
    int rg0 = row0 + w * 16 + hi * 8;
    int bb = rg0 >> 10, n0 = rg0 & 1023;
    if (region < 2) {
      unsigned short* dst = (unsigned short*)((region == 0) ? q : k);
      #pragma unroll
      for (int e = 0; e < 8; e++)
        dst[((size_t)(bb * H_ + h) * N_ + n0 + e) * DH_ + d] =
            f2bf(acc[nb][e] * scale);
    } else {
      // vT[b][h][d][n]: lane holds 8 consecutive n at fixed d -> one 16B store
      __align__(16) unsigned int pk[4];
      #pragma unroll
      for (int e = 0; e < 4; e++)
        pk[e] = (unsigned)f2bf(acc[nb][2 * e]) |
                ((unsigned)f2bf(acc[nb][2 * e + 1]) << 16);
      unsigned int* dst = (unsigned int*)((unsigned short*)vT +
          ((size_t)(bb * H_ + h) * DH_ + d) * N_ + n0);
      dst[0] = pk[0]; dst[1] = pk[1]; dst[2] = pk[2]; dst[3] = pk[3];
    }
  }
}

// ---------------------------------------------- K2: fused talking-heads ----
// One workgroup per (batch, 16-row i-tile); 12 waves = 12 heads.
// Two-pass flash: pass1 -> per-row (m,l) after mix_pre; pass2 recomputes
// scores, normalizes, applies mix_post across heads, WMMA attn@v.
__global__ __launch_bounds__(384) void k_attn(
    const __bf16* __restrict__ q, const __bf16* __restrict__ k,
    const __bf16* __restrict__ vT, const float* __restrict__ mix_pre,
    const float* __restrict__ mix_post, __bf16* __restrict__ o) {
  __shared__ __align__(16) float  s_xch[H_ * 16 * 32];   // 24 KB head-exchange
  __shared__ __align__(16) __bf16 s_stg[H_ * 16 * 40];   // 15 KB attn staging
  const int t = threadIdx.x, lane = t & 31, g = t >> 5;  // wave == head
  const int hi = lane >> 4, ln = lane & 15, Mb = hi * 8;
  const int bb = blockIdx.y, i0 = blockIdx.x * 16;

  float mp[H_], mq[H_];
  #pragma unroll
  for (int hh = 0; hh < H_; hh++) {
    mp[hh] = mix_pre[hh * H_ + g];
    mq[hh] = mix_post[hh * H_ + g];
  }

  const __bf16* qb = q + ((size_t)(bb * H_ + g) * N_ + i0) * DH_;
  const v16bf qa0 = load_a16(qb, DH_, lane);
  const v16bf qa1 = load_a16(qb + 32, DH_, lane);
  const __bf16* kh = k  + (size_t)(bb * H_ + g) * N_ * DH_;
  const __bf16* vh = vT + (size_t)(bb * H_ + g) * DH_ * N_;

  float m_run[8], l_run[8];
  #pragma unroll
  for (int e = 0; e < 8; e++) { m_run[e] = -1e30f; l_run[e] = 0.0f; }

  // ---------- pass 1: softmax stats ----------
  for (int j0 = 0; j0 < N_; j0 += 32) {
    v8f s0 = vzero8(), s1 = vzero8();
    const __bf16* kb0 = kh + (size_t)j0 * DH_;
    s0 = wmma_bf16(qa0, load_b16(kb0, DH_, lane), s0);
    s0 = wmma_bf16(qa1, load_b16(kb0 + 32, DH_, lane), s0);
    const __bf16* kb1 = kb0 + 16 * DH_;
    s1 = wmma_bf16(qa0, load_b16(kb1, DH_, lane), s1);
    s1 = wmma_bf16(qa1, load_b16(kb1 + 32, DH_, lane), s1);

    __syncthreads();
    #pragma unroll
    for (int e = 0; e < 8; e++) {
      s_xch[g * 512 + (Mb + e) * 32 + ln]      = s0[e];
      s_xch[g * 512 + (Mb + e) * 32 + 16 + ln] = s1[e];
    }
    __syncthreads();

    float sp0[8], sp1[8];
    #pragma unroll
    for (int e = 0; e < 8; e++) { sp0[e] = 0.f; sp1[e] = 0.f; }
    #pragma unroll
    for (int hh = 0; hh < H_; hh++) {
      float wgt = mp[hh];
      const float* sp = &s_xch[hh * 512 + Mb * 32 + ln];
      #pragma unroll
      for (int e = 0; e < 8; e++) {
        sp0[e] += wgt * sp[e * 32];
        sp1[e] += wgt * sp[e * 32 + 16];
      }
    }
    #pragma unroll
    for (int e = 0; e < 8; e++) {
      float mx = fmaxf(sp0[e], sp1[e]);
      mx = fmaxf(mx, __shfl_xor(mx, 1, 32));
      mx = fmaxf(mx, __shfl_xor(mx, 2, 32));
      mx = fmaxf(mx, __shfl_xor(mx, 4, 32));
      mx = fmaxf(mx, __shfl_xor(mx, 8, 32));
      float mnew = fmaxf(m_run[e], mx);
      float es = __expf(sp0[e] - mnew) + __expf(sp1[e] - mnew);
      es += __shfl_xor(es, 1, 32);
      es += __shfl_xor(es, 2, 32);
      es += __shfl_xor(es, 4, 32);
      es += __shfl_xor(es, 8, 32);
      l_run[e] = l_run[e] * __expf(m_run[e] - mnew) + es;
      m_run[e] = mnew;
    }
  }
  float inv_l[8];
  #pragma unroll
  for (int e = 0; e < 8; e++) inv_l[e] = 1.0f / l_run[e];

  // ---------- pass 2: recompute, normalize, mix_post, attn @ v ----------
  v8f oa[4];
  #pragma unroll
  for (int i = 0; i < 4; i++) oa[i] = vzero8();

  for (int j0 = 0; j0 < N_; j0 += 32) {
    v8f s0 = vzero8(), s1 = vzero8();
    const __bf16* kb0 = kh + (size_t)j0 * DH_;
    s0 = wmma_bf16(qa0, load_b16(kb0, DH_, lane), s0);
    s0 = wmma_bf16(qa1, load_b16(kb0 + 32, DH_, lane), s0);
    const __bf16* kb1 = kb0 + 16 * DH_;
    s1 = wmma_bf16(qa0, load_b16(kb1, DH_, lane), s1);
    s1 = wmma_bf16(qa1, load_b16(kb1 + 32, DH_, lane), s1);

    __syncthreads();
    #pragma unroll
    for (int e = 0; e < 8; e++) {
      s_xch[g * 512 + (Mb + e) * 32 + ln]      = s0[e];
      s_xch[g * 512 + (Mb + e) * 32 + 16 + ln] = s1[e];
    }
    __syncthreads();

    float a0[8], a1[8];
    #pragma unroll
    for (int e = 0; e < 8; e++) { a0[e] = 0.f; a1[e] = 0.f; }
    #pragma unroll
    for (int hh = 0; hh < H_; hh++) {
      float wgt = mp[hh];
      const float* sp = &s_xch[hh * 512 + Mb * 32 + ln];
      #pragma unroll
      for (int e = 0; e < 8; e++) {
        a0[e] += wgt * sp[e * 32];
        a1[e] += wgt * sp[e * 32 + 16];
      }
    }
    #pragma unroll
    for (int e = 0; e < 8; e++) {
      a0[e] = __expf(a0[e] - m_run[e]) * inv_l[e];
      a1[e] = __expf(a1[e] - m_run[e]) * inv_l[e];
    }

    __syncthreads();
    #pragma unroll
    for (int e = 0; e < 8; e++) {
      s_xch[g * 512 + (Mb + e) * 32 + ln]      = a0[e];
      s_xch[g * 512 + (Mb + e) * 32 + 16 + ln] = a1[e];
    }
    __syncthreads();

    float p0[8], p1[8];
    #pragma unroll
    for (int e = 0; e < 8; e++) { p0[e] = 0.f; p1[e] = 0.f; }
    #pragma unroll
    for (int hh = 0; hh < H_; hh++) {
      float wgt = mq[hh];
      const float* sp = &s_xch[hh * 512 + Mb * 32 + ln];
      #pragma unroll
      for (int e = 0; e < 8; e++) {
        p0[e] += wgt * sp[e * 32];
        p1[e] += wgt * sp[e * 32 + 16];
      }
    }
    // stage own mixed-attn tile (bf16, A-layout source) — wave-private
    unsigned short* st = (unsigned short*)&s_stg[g * 640];
    #pragma unroll
    for (int e = 0; e < 8; e++) {
      st[(Mb + e) * 40 + ln]      = f2bf(p0[e]);
      st[(Mb + e) * 40 + 16 + ln] = f2bf(p1[e]);
    }
    v16bf af = load_a16(&s_stg[g * 640], 40, lane);
    #pragma unroll
    for (int nb = 0; nb < 4; nb++) {
      const __bf16* vb = vh + (size_t)(nb * 16) * N_ + j0;
      oa[nb] = wmma_bf16(af, load_b16(vb, N_, lane), oa[nb]);
    }
  }

  // o[b][n][g*64 + d] bf16
  unsigned short* ob = (unsigned short*)o +
      ((size_t)bb * N_ + i0 + Mb) * INNER_ + g * 64;
  #pragma unroll
  for (int nb = 0; nb < 4; nb++)
    #pragma unroll
    for (int e = 0; e < 8; e++)
      ob[(size_t)e * INNER_ + nb * 16 + ln] = f2bf(oa[nb][e]);
}

// -------------------------------------------------- K3: output project ----
__global__ __launch_bounds__(256) void k_out(
    const __bf16* __restrict__ ob, const __bf16* __restrict__ WoT,
    const float* __restrict__ bo, float* __restrict__ out) {
  __shared__ __align__(16) __bf16 ldsX[2][128 * 40];
  __shared__ __align__(16) __bf16 ldsW[2][64 * 40];
  const int t = threadIdx.x, lane = t & 31, w = t >> 5;
  const int jl0 = blockIdx.x * 64;
  const int row0 = blockIdx.y * 128;

  const int xr = t >> 1, xc = (t & 1) * 16;
  const int wj = t >> 2, wc = (t & 3) * 8;
  const __bf16* xp = ob  + (size_t)(row0 + xr) * INNER_ + xc;
  const __bf16* wp = WoT + (size_t)(jl0 + wj) * INNER_ + wc;
  const unsigned lx[2] = { lds_off(&ldsX[0][xr * 40 + xc]),
                           lds_off(&ldsX[1][xr * 40 + xc]) };
  const unsigned lw[2] = { lds_off(&ldsW[0][wj * 40 + wc]),
                           lds_off(&ldsW[1][wj * 40 + wc]) };

  v8f acc[4];
  #pragma unroll
  for (int i = 0; i < 4; i++) acc[i] = vzero8();

  async_ld_b128(lx[0],      xp);
  async_ld_b128(lx[0] + 16, xp + 8);
  async_ld_b128(lw[0],      wp);

  for (int kc = 0; kc < INNER_; kc += 32) {
    const int cur = (kc >> 5) & 1;
    if (kc + 32 < INNER_) {
      const int nb = cur ^ 1;
      async_ld_b128(lx[nb],      xp + kc + 32);
      async_ld_b128(lx[nb] + 16, xp + kc + 40);
      async_ld_b128(lw[nb],      wp + kc + 32);
      asm volatile("s_wait_asynccnt 0x3" ::: "memory");
    } else {
      asm volatile("s_wait_asynccnt 0x0" ::: "memory");
    }
    __syncthreads();
    v16bf a = load_a16(&ldsX[cur][(w * 16) * 40], 40, lane);
    #pragma unroll
    for (int nb = 0; nb < 4; nb++) {
      v16bf b = load_b16(&ldsW[cur][(nb * 16) * 40], 40, lane);
      acc[nb] = wmma_bf16(a, b, acc[nb]);
    }
    __syncthreads();
  }

  const int hi = lane >> 4, ln = lane & 15;
  #pragma unroll
  for (int nb = 0; nb < 4; nb++) {
    int j = jl0 + nb * 16 + ln;
    float bias = bo[j];
    #pragma unroll
    for (int e = 0; e < 8; e++) {
      int rg = row0 + w * 16 + hi * 8 + e;
      out[(size_t)rg * DIM_ + j] = acc[nb][e] + bias;
    }
  }
}

// ------------------------------------------------------------- launcher ----
extern "C" void kernel_launch(void* const* d_in, const int* in_sizes, int n_in,
                              void* d_out, int out_size, void* d_ws,
                              size_t ws_size, hipStream_t stream) {
  const float* x        = (const float*)d_in[0];
  const float* Wq       = (const float*)d_in[1];
  const float* Wkv      = (const float*)d_in[2];
  const float* mix_pre  = (const float*)d_in[3];
  const float* mix_post = (const float*)d_in[4];
  const float* Wo       = (const float*)d_in[5];
  const float* bo       = (const float*)d_in[6];
  float* out = (float*)d_out;

  char* ws = (char*)d_ws;
  size_t off = 0;
  auto carve = [&](size_t bytes) {
    char* p = ws + off;
    off += (bytes + 255) & ~(size_t)255;
    return p;
  };
  __bf16* xb  = (__bf16*)carve((size_t)B_ * N_ * DIM_ * 2);
  __bf16* WqT = (__bf16*)carve((size_t)DIM_ * INNER_ * 2);
  __bf16* WkT = (__bf16*)carve((size_t)DIM_ * INNER_ * 2);
  __bf16* WvT = (__bf16*)carve((size_t)DIM_ * INNER_ * 2);
  __bf16* WoT = (__bf16*)carve((size_t)INNER_ * DIM_ * 2);
  __bf16* qb  = (__bf16*)carve((size_t)B_ * H_ * N_ * DH_ * 2);
  __bf16* kb  = (__bf16*)carve((size_t)B_ * H_ * N_ * DH_ * 2);
  __bf16* vTb = (__bf16*)carve((size_t)B_ * H_ * N_ * DH_ * 2);
  __bf16* obf = (__bf16*)carve((size_t)B_ * N_ * INNER_ * 2);
  (void)ws_size; (void)in_sizes; (void)n_in; (void)out_size;

  const int nX = B_ * N_ * DIM_;
  k_cvt_f32_bf16<<<(nX + 255) / 256, 256, 0, stream>>>(x, xb, nX);
  const int nW = DIM_ * INNER_;
  k_transpose_cvt<<<(nW + 255) / 256, 256, 0, stream>>>(Wq,  WqT, DIM_, INNER_, INNER_, 0);
  k_transpose_cvt<<<(nW + 255) / 256, 256, 0, stream>>>(Wkv, WkT, DIM_, INNER_, 2 * INNER_, 0);
  k_transpose_cvt<<<(nW + 255) / 256, 256, 0, stream>>>(Wkv, WvT, DIM_, INNER_, 2 * INNER_, INNER_);
  k_transpose_cvt<<<(nW + 255) / 256, 256, 0, stream>>>(Wo,  WoT, INNER_, DIM_, DIM_, 0);

  k_qkv<<<dim3(36, 64), 256, 0, stream>>>(xb, WqT, WkT, WvT, qb, kb, vTb);
  k_attn<<<dim3(N_ / 16, B_), 384, 0, stream>>>(qb, kb, vTb, mix_pre, mix_post, obf);
  k_out<<<dim3(12, 64), 256, 0, stream>>>(obf, WoT, bo, out);
}
